// CausalDecayMemory_3865470566777
// MI455X (gfx1250) — matrix operand order repaired and checked by reading, not compiled, not run on hardware
//
#include <hip/hip_runtime.h>

// Problem constants (fixed by the reference)
#define BB 4
#define TB 4096
#define DD 512

typedef _Float16 half_t;
typedef _Float16 v16h __attribute__((ext_vector_type(16)));
typedef float    v8f  __attribute__((ext_vector_type(8)));

// Assemble a 16-half WMMA fragment from two 16-byte chunks.
// A-matrix frag: p1 = p0 + 8  (16 contiguous K values)
// B-matrix frag: p1 = p0 + 8  (16 contiguous K values)
__device__ __forceinline__ v16h frag2(const half_t* p0, const half_t* p1) {
  v16h r;
  ((uint4*)&r)[0] = *(const uint4*)p0;
  ((uint4*)&r)[1] = *(const uint4*)p1;
  return r;
}

__device__ __forceinline__ v8f wmma_f16(v16h a, v16h b, v8f c) {
  // D = A(16x32 f16) * B(32x16 f16) + C(16x16 f32)
  return __builtin_amdgcn_wmma_f32_16x16x32_f16(
      /*neg_a=*/false, a, /*neg_b=*/false, b,
      /*c_mod=*/(short)0, c, /*reuse_a=*/false, /*reuse_b=*/false);
}

// ---------------------------------------------------------------------------
// f32 -> f16 conversion
// ---------------------------------------------------------------------------
__global__ void f32_to_f16_kernel(const float* __restrict__ in,
                                  half_t* __restrict__ out, int n) {
  int i = blockIdx.x * blockDim.x + threadIdx.x;
  if (i < n) out[i] = (half_t)in[i];
}

// ---------------------------------------------------------------------------
// Y = A (Mx512 f16, row-major) @ W^T  where W is 512x512 f16 row-major.
// WMMA B-frag wants B[k][n] = W[n][k]  -> contiguous row reads of W.
// MODE 0: f16 out, row-major MxD          (q, k)
// MODE 1: f16 out, per-batch transposed   (v -> layout B x D x T)
// MODE 2: f32 out, row-major, * (*scale)  (final output)
// Block: 128 threads = 4 waves in 2x2, block tile 64x64, wave tile 32x32.
// ---------------------------------------------------------------------------
template <int MODE>
__global__ __launch_bounds__(128) void gemm_xw_kernel(
    const half_t* __restrict__ A, const half_t* __restrict__ W,
    void* __restrict__ out, const float* __restrict__ scale_ptr) {
  const int lane = threadIdx.x & 31;
  const int wid  = threadIdx.x >> 5;
  const int wm   = wid & 1;          // wave row (0..1)
  const int wn   = wid >> 1;         // wave col (0..1)
  const int hi   = lane >> 4;        // half-wave select
  const int ln   = lane & 15;
  const int kbA  = hi * 16;          // A-frag K base
  const int kbB  = hi * 16;          // B-frag K base
  const int m_blk = blockIdx.x * 64;
  const int n_blk = blockIdx.y * 64;

  v8f acc[2][2] = {};

  const half_t* arow0 = A + (size_t)(m_blk + wm * 32 + ln) * DD;
  const half_t* arow1 = arow0 + (size_t)16 * DD;
  const half_t* wrow0 = W + (size_t)(n_blk + wn * 32 + ln) * DD;
  const half_t* wrow1 = wrow0 + (size_t)16 * DD;

#pragma unroll 4
  for (int k0 = 0; k0 < DD; k0 += 32) {
    v16h a0 = frag2(arow0 + k0 + kbA, arow0 + k0 + kbA + 8);
    v16h a1 = frag2(arow1 + k0 + kbA, arow1 + k0 + kbA + 8);
    v16h b0 = frag2(wrow0 + k0 + kbB, wrow0 + k0 + kbB + 8);
    v16h b1 = frag2(wrow1 + k0 + kbB, wrow1 + k0 + kbB + 8);
    acc[0][0] = wmma_f16(a0, b0, acc[0][0]);
    acc[0][1] = wmma_f16(a0, b1, acc[0][1]);
    acc[1][0] = wmma_f16(a1, b0, acc[1][0]);
    acc[1][1] = wmma_f16(a1, b1, acc[1][1]);
  }

  const float scale = (MODE == 2) ? *scale_ptr : 1.0f;
  for (int mt = 0; mt < 2; ++mt)
    for (int nt = 0; nt < 2; ++nt)
      for (int r = 0; r < 8; ++r) {
        const int m = m_blk + wm * 32 + mt * 16 + r + hi * 8;
        const int n = n_blk + wn * 32 + nt * 16 + ln;
        const float c = acc[mt][nt][r];
        if (MODE == 0) {
          ((half_t*)out)[(size_t)m * DD + n] = (half_t)c;
        } else if (MODE == 1) {
          const int b = m >> 12;        // T == 4096
          const int ml = m & (TB - 1);
          ((half_t*)out)[(size_t)b * DD * TB + (size_t)n * TB + ml] = (half_t)c;
        } else {
          ((float*)out)[(size_t)m * DD + n] = c * scale;
        }
      }
}

// ---------------------------------------------------------------------------
// Attention with forward-looking exponential decay:
//   retrieved[t] = sum_{s>t} (q_t . k_s) * decay^(s-t-1) * v_s
// Flash-style: 32-row query tile per block (256 thr / 8 waves).
// Per 128-wide s-tile:
//   phase 1: S = Q K^T  (wave w owns s-cols [16w,16w+16), 2 M-tiles, K=512)
//   decay weights applied element-wise (geometric recurrence: one v_mul per
//   element per tile, exp2 only at init), P (f16) staged via LDS,
//   phase 2: acc += P V  (wave w owns d-cols [64w,64w+64), K=128)
// K and V(transposed) fragments stream straight from L2 (whole f16 working
// set is L2-resident on MI455X: 192MB L2).
// ---------------------------------------------------------------------------
__global__ __launch_bounds__(256) void attn_decay_kernel(
    const half_t* __restrict__ qh, const half_t* __restrict__ kh,
    const half_t* __restrict__ vT, half_t* __restrict__ rh,
    const float* __restrict__ decay_logit) {
  constexpr int PS = 136;  // P row stride in halves (272B = 17*16B: aligned, conflict-skewed)
  __shared__ __align__(16) half_t ldsP[32 * PS];

  const int lane = threadIdx.x & 31;
  const int wid  = threadIdx.x >> 5;   // 0..7
  const int bid  = blockIdx.x;         // 0 .. B*T/32-1
  const int b    = bid >> 7;           // 128 query tiles per batch
  const int t0   = (bid & 127) * 32;

  const int hi  = lane >> 4;
  const int ln  = lane & 15;
  const int kbA = hi * 16;
  const int kbB = hi * 16;

  const float dl  = *decay_logit;
  const float dec = 1.0f / (1.0f + __expf(-dl));
  const float l2d = __log2f(dec);
  const float d128 = exp2f(l2d * 128.0f);   // dec^128: per-s-tile weight update

  const half_t* qrow0 = qh + (size_t)(b * TB + t0 + ln) * DD;        // M-tile 0
  const half_t* qrow1 = qrow0 + (size_t)16 * DD;                     // M-tile 1
  const half_t* kbase = kh + (size_t)(b * TB) * DD;
  const half_t* vbase = vT + (size_t)b * DD * TB;

  // V rows owned by this wave (B-frag source for phase 2)
  const half_t* vrow0 = vbase + (size_t)(wid * 64 + 0 * 16 + ln) * TB;
  const half_t* vrow1 = vbase + (size_t)(wid * 64 + 1 * 16 + ln) * TB;
  const half_t* vrow2 = vbase + (size_t)(wid * 64 + 2 * 16 + ln) * TB;
  const half_t* vrow3 = vbase + (size_t)(wid * 64 + 3 * 16 + ln) * TB;

  v8f acc[2][4] = {};

  const int s_begin = t0 & ~127;
  // Unmasked decay weights dec^(srel - tl - 1) for this lane's 16 C-elements.
  // srel - tl - 1 >= -128 so values are bounded (<= dec^-128 ~ 512).
  // The (s > t) mask only matters on the first (diagonal) s-tile.
  const int srel0 = (s_begin - t0) + wid * 16 + ln;  // s - t0 for this lane's column
  float wA0[8], wA1[8];
#pragma unroll
  for (int r = 0; r < 8; ++r) {
    const int tl0 = r + hi * 8;
    const int tl1 = 16 + r + hi * 8;
    wA0[r] = exp2f(l2d * (float)(srel0 - tl0 - 1));
    wA1[r] = exp2f(l2d * (float)(srel0 - tl1 - 1));
  }

  for (int s0 = s_begin; s0 < TB; s0 += 128) {
    // ---------------- phase 1: scores ----------------
    const half_t* krow = kbase + (size_t)(s0 + wid * 16 + ln) * DD;
    if (s0 + 128 < TB)  // gfx1250 global_prefetch_b8 for next K tile
      __builtin_prefetch(krow + (size_t)128 * DD, 0, 1);
    // pull this tile's V rows toward the caches while scores compute
    __builtin_prefetch(vrow0 + s0, 0, 1);
    __builtin_prefetch(vrow1 + s0, 0, 1);
    __builtin_prefetch(vrow2 + s0, 0, 1);
    __builtin_prefetch(vrow3 + s0, 0, 1);

    v8f sacc0 = {}, sacc1 = {};
#pragma unroll 4
    for (int d0 = 0; d0 < DD; d0 += 32) {
      v16h bfr = frag2(krow + d0 + kbB, krow + d0 + kbB + 8);
      v16h a0  = frag2(qrow0 + d0 + kbA, qrow0 + d0 + kbA + 8);
      v16h a1  = frag2(qrow1 + d0 + kbA, qrow1 + d0 + kbA + 8);
      sacc0 = wmma_f16(a0, bfr, sacc0);
      sacc1 = wmma_f16(a1, bfr, sacc1);
    }

    // Decay weights + stage P into LDS (f16). Steady state: 2 v_mul per element.
    const int scol = wid * 16 + ln;
    const bool diag = (s0 == s_begin);
#pragma unroll
    for (int r = 0; r < 8; ++r) {
      const int tl0 = r + hi * 8;
      const int tl1 = 16 + r + hi * 8;
      float w0 = wA0[r];
      float w1 = wA1[r];
      if (diag) {  // apply s > t mask (only possible on the diagonal tile)
        w0 = (srel0 - tl0 > 0) ? w0 : 0.0f;
        w1 = (srel0 - tl1 > 0) ? w1 : 0.0f;
      }
      ldsP[tl0 * PS + scol] = (half_t)(sacc0[r] * w0);
      ldsP[tl1 * PS + scol] = (half_t)(sacc1[r] * w1);
      wA0[r] *= d128;
      wA1[r] *= d128;
    }
    __syncthreads();

    // ---------------- phase 2: acc += P @ V ----------------
#pragma unroll
    for (int kc = 0; kc < 128; kc += 32) {
      v16h a0 = frag2(&ldsP[ln * PS + kc + kbA], &ldsP[ln * PS + kc + kbA + 8]);
      v16h a1 = frag2(&ldsP[(16 + ln) * PS + kc + kbA], &ldsP[(16 + ln) * PS + kc + kbA + 8]);
      v16h b0 = frag2(vrow0 + s0 + kc + kbB, vrow0 + s0 + kc + kbB + 8);
      v16h b1 = frag2(vrow1 + s0 + kc + kbB, vrow1 + s0 + kc + kbB + 8);
      v16h b2 = frag2(vrow2 + s0 + kc + kbB, vrow2 + s0 + kc + kbB + 8);
      v16h b3 = frag2(vrow3 + s0 + kc + kbB, vrow3 + s0 + kc + kbB + 8);
      acc[0][0] = wmma_f16(a0, b0, acc[0][0]);
      acc[1][0] = wmma_f16(a1, b0, acc[1][0]);
      acc[0][1] = wmma_f16(a0, b1, acc[0][1]);
      acc[1][1] = wmma_f16(a1, b1, acc[1][1]);
      acc[0][2] = wmma_f16(a0, b2, acc[0][2]);
      acc[1][2] = wmma_f16(a1, b2, acc[1][2]);
      acc[0][3] = wmma_f16(a0, b3, acc[0][3]);
      acc[1][3] = wmma_f16(a1, b3, acc[1][3]);
    }
    __syncthreads();
  }

  // Write retrieved (f16, row-major) for the final Wo GEMM
  for (int mt = 0; mt < 2; ++mt)
    for (int nt = 0; nt < 4; ++nt)
      for (int r = 0; r < 8; ++r) {
        const int trow = t0 + mt * 16 + r + hi * 8;
        const int ncol = wid * 64 + nt * 16 + ln;
        rh[(size_t)(b * TB + trow) * DD + ncol] = (half_t)acc[mt][nt][r];
      }
}

// ---------------------------------------------------------------------------
// Launcher
// inputs: x, Wq, Wk, Wv, Wo, decay_logit, out_scale (all f32)
// output: (B,T,D) f32
// ---------------------------------------------------------------------------
extern "C" void kernel_launch(void* const* d_in, const int* in_sizes, int n_in,
                              void* d_out, int out_size, void* d_ws, size_t ws_size,
                              hipStream_t stream) {
  (void)in_sizes; (void)n_in; (void)out_size; (void)ws_size;
  const float* x           = (const float*)d_in[0];
  const float* Wq          = (const float*)d_in[1];
  const float* Wk          = (const float*)d_in[2];
  const float* Wv          = (const float*)d_in[3];
  const float* Wo          = (const float*)d_in[4];
  const float* decay_logit = (const float*)d_in[5];
  const float* out_scale   = (const float*)d_in[6];

  const size_t NXD = (size_t)BB * TB * DD;  // 8,388,608
  const size_t WSZ = (size_t)DD * DD;       // 262,144

  half_t* ws  = (half_t*)d_ws;
  half_t* xh  = ws;
  half_t* qh  = ws + 1 * NXD;
  half_t* khp = ws + 2 * NXD;
  half_t* vT  = ws + 3 * NXD;   // transposed: B x D x T
  half_t* rh  = ws + 4 * NXD;
  half_t* Wqh = ws + 5 * NXD;
  half_t* Wkh = Wqh + 1 * WSZ;
  half_t* Wvh = Wqh + 2 * WSZ;
  half_t* Woh = Wqh + 3 * WSZ;

  const int nx = (int)NXD, nw = (int)WSZ;
  f32_to_f16_kernel<<<(nx + 255) / 256, 256, 0, stream>>>(x, xh, nx);
  f32_to_f16_kernel<<<(nw + 255) / 256, 256, 0, stream>>>(Wq, Wqh, nw);
  f32_to_f16_kernel<<<(nw + 255) / 256, 256, 0, stream>>>(Wk, Wkh, nw);
  f32_to_f16_kernel<<<(nw + 255) / 256, 256, 0, stream>>>(Wv, Wvh, nw);
  f32_to_f16_kernel<<<(nw + 255) / 256, 256, 0, stream>>>(Wo, Woh, nw);

  dim3 gg(BB * TB / 64, DD / 64);  // 256 x 8 blocks
  gemm_xw_kernel<0><<<gg, 128, 0, stream>>>(xh, Wqh, qh, nullptr);
  gemm_xw_kernel<0><<<gg, 128, 0, stream>>>(xh, Wkh, khp, nullptr);
  gemm_xw_kernel<1><<<gg, 128, 0, stream>>>(xh, Wvh, vT, nullptr);

  attn_decay_kernel<<<BB * TB / 32, 256, 0, stream>>>(qh, khp, vT, rh, decay_logit);

  gemm_xw_kernel<2><<<gg, 128, 0, stream>>>(rh, Woh, d_out, out_scale);
}
